// Decoder_76888504533579
// MI455X (gfx1250) — matrix operand order repaired and checked by reading, not compiled, not run on hardware
//
#include <hip/hip_runtime.h>
#include <hip/hip_bf16.h>
#include <math.h>

// Problem constants (match reference)
#define B_   16
#define T_   32
#define S_   256
#define KK   8
#define LL   64
#define VV   32000
#define VX_  32050
#define HH   512
#define EE   512
#define NEGBIG (-1e10f)

typedef __attribute__((ext_vector_type(16))) _Float16 v16h;
typedef __attribute__((ext_vector_type(8)))  _Float16 v8h;
typedef __attribute__((ext_vector_type(8)))  float    v8f;

// ---------------------------------------------------------------------------
// block-wide reduction for 256-thread blocks (8 wave32 waves)
// ---------------------------------------------------------------------------
__device__ inline float block_reduce_256(float v, float* red8, bool isMax) {
  const int tid  = threadIdx.x;
  const int lane = tid & 31, wid = tid >> 5;
  #pragma unroll
  for (int off = 16; off > 0; off >>= 1) {
    float o = __shfl_down(v, off, 32);
    v = isMax ? fmaxf(v, o) : (v + o);
  }
  if (lane == 0) red8[wid] = v;
  __syncthreads();
  if (tid == 0) {
    float r = red8[0];
    for (int i = 1; i < 8; ++i) r = isMax ? fmaxf(r, red8[i]) : (r + red8[i]);
    red8[0] = r;
  }
  __syncthreads();
  float r = red8[0];
  __syncthreads();
  return r;
}

// ---------------------------------------------------------------------------
// f32 -> f16 conversion (one-time prep; converted tensors then live in L2)
// ---------------------------------------------------------------------------
__global__ void cvt_f16(const float* __restrict__ src, _Float16* __restrict__ dst, size_t n) {
  size_t i = (size_t)blockIdx.x * blockDim.x + threadIdx.x;
  size_t stride = (size_t)gridDim.x * blockDim.x;
  for (; i < n; i += stride) dst[i] = (_Float16)src[i];
}

// ---------------------------------------------------------------------------
// WMMA GEMM: out[M,N] = A[M,K] @ W[N,K]^T + bias[N]
// A and W both f16 row-major; f32 accumulate. One wave per 16x16 output tile,
// K unrolled by 64 (2 x v_wmma_f32_16x16x32_f16 per iteration). Fragment
// loads are pure global_load_b128; weight stream prefetched one chunk ahead.
// M,N multiples of 16; K multiple of 64. Wave-uniform early-exit keeps EXEC
// all-ones for WMMA (ISA 7.12 restriction).
// ---------------------------------------------------------------------------
__global__ void gemm_h(const _Float16* __restrict__ A,
                       const _Float16* __restrict__ W,
                       const float* __restrict__ bias,
                       float*    __restrict__ outf,   // optional f32 output
                       _Float16* __restrict__ outh,   // optional f16 output
                       int M, int N, int Kd) {
  const int lane   = threadIdx.x & 31;
  const int wave   = (blockIdx.x * blockDim.x + threadIdx.x) >> 5;
  const int tilesN = N >> 4;
  const int nTiles = (M >> 4) * tilesN;
  if (wave >= nTiles) return;
  const int tm = wave / tilesN;
  const int tn = wave - tm * tilesN;
  const int m0 = tm << 4, n0 = tn << 4;

  const int hiA = (lane >> 4) ? 16 : 0;   // A: lanes 16-31 hold K+16 halves
  const int hiB = (lane >> 4) ? 16 : 0;   // B: lanes 16-31 hold K+16 halves
  const _Float16* arow = A + (size_t)(m0 + (lane & 15)) * Kd + hiA;
  const _Float16* wrow = W + (size_t)(n0 + (lane & 15)) * Kd + hiB;

  v8f acc = {};
  for (int k0 = 0; k0 < Kd; k0 += 64) {
    // prefetch next 64-wide K chunk of the weight row (global_prefetch_b8)
    __builtin_prefetch((const void*)(wrow + k0 + 64), 0, 1);
    // ---- chunk 0 (K = k0..k0+31) ----
    v8h a0lo = *reinterpret_cast<const v8h*>(arow + k0);
    v8h a0hi = *reinterpret_cast<const v8h*>(arow + k0 + 8);
    v16h a0  = __builtin_shufflevector(a0lo, a0hi, 0,1,2,3,4,5,6,7,8,9,10,11,12,13,14,15);
    v16h b0  = *reinterpret_cast<const v16h*>(wrow + k0);
    // ---- chunk 1 (K = k0+32..k0+63) ----
    v8h a1lo = *reinterpret_cast<const v8h*>(arow + k0 + 32);
    v8h a1hi = *reinterpret_cast<const v8h*>(arow + k0 + 40);
    v16h a1  = __builtin_shufflevector(a1lo, a1hi, 0,1,2,3,4,5,6,7,8,9,10,11,12,13,14,15);
    v16h b1  = *reinterpret_cast<const v16h*>(wrow + k0 + 32);

    acc = __builtin_amdgcn_wmma_f32_16x16x32_f16(false, a0, false, b0,
                                                 (short)0, acc, false, false);
    acc = __builtin_amdgcn_wmma_f32_16x16x32_f16(false, a1, false, b1,
                                                 (short)0, acc, false, false);
  }
  const int   nn    = n0 + (lane & 15);
  const float bb    = bias ? bias[nn] : 0.0f;
  const int   mBase = m0 + ((lane >> 4) ? 8 : 0);
  if (outf) {
    #pragma unroll
    for (int r = 0; r < 8; ++r)
      outf[(size_t)(mBase + r) * N + nn] = acc[r] + bb;
  }
  if (outh) {
    #pragma unroll
    for (int r = 0; r < 8; ++r)
      outh[(size_t)(mBase + r) * N + nn] = (_Float16)(acc[r] + bb);
  }
}

// ---------------------------------------------------------------------------
// Additive attention over src (S=256) and kg (K=8). One block per batch row.
// ---------------------------------------------------------------------------
__global__ void attend_kernel(const float* __restrict__ h,
                              const float* __restrict__ Wq_c, const float* __restrict__ Vw_c,
                              const float* __restrict__ bV_c,
                              const float* __restrict__ pv_src, const float* __restrict__ src_out,
                              const float* __restrict__ src_mask,
                              const float* __restrict__ Wq_k, const float* __restrict__ Vw_k,
                              const float* __restrict__ bV_k,
                              const float* __restrict__ pv_kg, const float* __restrict__ kg_hid,
                              const float* __restrict__ kg_mask,
                              float* __restrict__ c_t, float* __restrict__ k_t,
                              float* __restrict__ k_w) {
  __shared__ float hs[HH];
  __shared__ float hq[HH];
  __shared__ float ew[S_];
  __shared__ float red[8];
  const int b = blockIdx.x;
  const int tid = threadIdx.x;              // 256 threads
  const int lane = tid & 31, wid = tid >> 5;

  for (int i = tid; i < HH; i += 256) hs[i] = h[b * HH + i];
  __syncthreads();

  // ---- src attention ----
  for (int j = tid; j < HH; j += 256) {     // hq = h @ Wq_c^T
    const float* wr = Wq_c + (size_t)j * HH;
    float acc = 0.f;
    for (int x = 0; x < HH; ++x) acc += hs[x] * wr[x];
    hq[j] = acc;
  }
  __syncthreads();
  for (int s = wid; s < S_; s += 8) {       // scores (one wave per s)
    const float* pv = pv_src + ((size_t)b * S_ + s) * HH;
    float p = 0.f;
    for (int x = lane; x < HH; x += 32) p += Vw_c[x] * tanhf(hq[x] + pv[x]);
    #pragma unroll
    for (int off = 16; off > 0; off >>= 1) p += __shfl_down(p, off, 32);
    if (lane == 0) {
      float bias = (src_mask[b * S_ + s] == 0.f) ? NEGBIG : 0.f;
      ew[s] = p + bV_c[0] + bias;
    }
  }
  __syncthreads();
  float m = -1e30f;
  for (int s = tid; s < S_; s += 256) m = fmaxf(m, ew[s]);
  m = block_reduce_256(m, red, true);
  float ssum = 0.f;
  for (int s = tid; s < S_; s += 256) { float v = __expf(ew[s] - m); ew[s] = v; ssum += v; }
  float tot = block_reduce_256(ssum, red, false);
  float inv = 1.f / tot;
  for (int s = tid; s < S_; s += 256) ew[s] *= inv;
  __syncthreads();
  for (int x = tid; x < HH; x += 256) {     // c_t = w @ src_out
    float acc = 0.f;
    for (int s = 0; s < S_; ++s) acc += ew[s] * src_out[((size_t)b * S_ + s) * HH + x];
    c_t[b * HH + x] = acc;
  }
  __syncthreads();

  // ---- kg attention (K=8) ----
  for (int j = tid; j < HH; j += 256) {
    const float* wr = Wq_k + (size_t)j * HH;
    float acc = 0.f;
    for (int x = 0; x < HH; ++x) acc += hs[x] * wr[x];
    hq[j] = acc;
  }
  __syncthreads();
  if (wid < KK) {
    const int s = wid;
    const float* pv = pv_kg + ((size_t)b * KK + s) * HH;
    float p = 0.f;
    for (int x = lane; x < HH; x += 32) p += Vw_k[x] * tanhf(hq[x] + pv[x]);
    #pragma unroll
    for (int off = 16; off > 0; off >>= 1) p += __shfl_down(p, off, 32);
    if (lane == 0) {
      float bias = (kg_mask[b * KK + s] == 0.f) ? NEGBIG : 0.f;
      ew[s] = p + bV_k[0] + bias;
    }
  }
  __syncthreads();
  if (tid == 0) {
    float mm = ew[0];
    for (int i = 1; i < KK; ++i) mm = fmaxf(mm, ew[i]);
    float s = 0.f;
    for (int i = 0; i < KK; ++i) { ew[i] = __expf(ew[i] - mm); s += ew[i]; }
    float iv = 1.f / s;
    for (int i = 0; i < KK; ++i) ew[i] *= iv;
  }
  __syncthreads();
  for (int x = tid; x < HH; x += 256) {
    float acc = 0.f;
    for (int s = 0; s < KK; ++s) acc += ew[s] * kg_hid[((size_t)b * KK + s) * HH + x];
    k_t[b * HH + x] = acc;
  }
  if (tid < KK) k_w[b * KK + tid] = ew[tid];
}

// ---------------------------------------------------------------------------
// Build GRU input x = [embed(y_t), c_t, k_t]: f32 (for ptr dot) + f16 (GEMM A)
// ---------------------------------------------------------------------------
__global__ void build_x(const int* __restrict__ tgt, const float* __restrict__ emb,
                        const float* __restrict__ c_t, const float* __restrict__ k_t,
                        float* __restrict__ xb, _Float16* __restrict__ xh, int t) {
  const int b = blockIdx.x, tid = threadIdx.x;
  const int row = tgt[b * T_ + t];
  float*     xr = xb + (size_t)b * 1536;
  _Float16*  xrh = xh + (size_t)b * 1536;
  for (int i = tid; i < HH; i += 256) {
    float y = emb[(size_t)row * EE + i];
    float c = c_t[b * HH + i];
    float k = k_t[b * HH + i];
    xr[i] = y;            xrh[i] = (_Float16)y;
    xr[HH + i] = c;       xrh[HH + i] = (_Float16)c;
    xr[2 * HH + i] = k;   xrh[2 * HH + i] = (_Float16)k;
  }
}

// ---------------------------------------------------------------------------
// GRU gate combine -> s_t (updates h f32+f16), s_cat (f16), ptr
// ---------------------------------------------------------------------------
__global__ void gru_combine(const float* __restrict__ gi, const float* __restrict__ gh,
                            float* __restrict__ hbuf, _Float16* __restrict__ hbuf_h,
                            const float* __restrict__ c_t, const float* __restrict__ k_t,
                            const float* __restrict__ xb, const float* __restrict__ kg_fus,
                            const float* __restrict__ Wptr, const float* __restrict__ bptr,
                            _Float16* __restrict__ sbuf_h, _Float16* __restrict__ scat_h,
                            float* __restrict__ ptrbuf) {
  __shared__ float s_sh[HH];
  __shared__ float red[8];
  const int b = blockIdx.x, tid = threadIdx.x;   // 256 threads
  const float* gib = gi + (size_t)b * 1536;
  const float* ghb = gh + (size_t)b * 1536;
  for (int i = tid; i < HH; i += 256) {
    float r = 1.f / (1.f + __expf(-(gib[i] + ghb[i])));
    float z = 1.f / (1.f + __expf(-(gib[HH + i] + ghb[HH + i])));
    float n = tanhf(gib[2 * HH + i] + r * ghb[2 * HH + i]);
    float hold = hbuf[b * HH + i];
    s_sh[i] = (1.f - z) * n + z * hold;
  }
  __syncthreads();
  for (int i = tid; i < HH; i += 256) {
    float s = s_sh[i];
    hbuf[b * HH + i]   = s;                      // recurrent state (f32)
    hbuf_h[b * HH + i] = (_Float16)s;            // recurrent state (f16, GEMM A)
    sbuf_h[b * HH + i] = (_Float16)s;
    _Float16* sc = scat_h + (size_t)b * 2048;
    sc[i]           = (_Float16)s;
    sc[HH + i]      = (_Float16)c_t[b * HH + i];
    sc[2 * HH + i]  = (_Float16)k_t[b * HH + i];
    sc[3 * HH + i]  = (_Float16)kg_fus[b * HH + i];
  }
  // ptr = sigmoid(Wptr . [c, s, y, kf, k] + bptr)
  float p = 0.f;
  for (int i = tid; i < HH; i += 256) {
    p += Wptr[i]          * c_t[b * HH + i];
    p += Wptr[HH + i]     * s_sh[i];
    p += Wptr[2 * HH + i] * xb[(size_t)b * 1536 + i];
    p += Wptr[3 * HH + i] * kg_fus[b * HH + i];
    p += Wptr[4 * HH + i] * k_t[b * HH + i];
  }
  float tot = block_reduce_256(p, red, false);
  if (tid == 0) ptrbuf[b] = 1.f / (1.f + __expf(-(tot + bptr[0])));
}

// ---------------------------------------------------------------------------
// Copy distribution: softmax_l(q . kg_out[b,k,l]) * k_w * (1-ptr)
// ---------------------------------------------------------------------------
__global__ void copy_scores(const float* __restrict__ qbuf, const _Float16* __restrict__ kgout_h,
                            const float* __restrict__ kg_pad, const float* __restrict__ k_w,
                            const float* __restrict__ ptrbuf, float* __restrict__ val) {
  __shared__ float q_sh[HH];
  __shared__ float sc[LL];
  const int bk = blockIdx.x;       // 0..127
  const int b = bk >> 3;
  const int tid = threadIdx.x;     // 64 threads
  for (int i = tid; i < HH; i += 64) q_sh[i] = qbuf[b * HH + i];
  __syncthreads();
  const _Float16* row = kgout_h + ((size_t)bk * LL + tid) * HH;
  float acc = 0.f;
  for (int x = 0; x < HH; ++x) acc += q_sh[x] * (float)row[x];
  float bias = (kg_pad[bk * LL + tid] == 0.f) ? NEGBIG : 0.f;
  sc[tid] = acc + bias;
  __syncthreads();
  if (tid == 0) {
    float mm = sc[0];
    for (int i = 1; i < LL; ++i) mm = fmaxf(mm, sc[i]);
    float s = 0.f;
    for (int i = 0; i < LL; ++i) { sc[i] = __expf(sc[i] - mm); s += sc[i]; }
    float iv = 1.f / s;
    for (int i = 0; i < LL; ++i) sc[i] *= iv;
  }
  __syncthreads();
  val[(size_t)b * (KK * LL) + (bk & 7) * LL + tid] = sc[tid] * k_w[bk] * (1.f - ptrbuf[b]);
}

// ---------------------------------------------------------------------------
// Per-row softmax stats over V=32000 logits
// ---------------------------------------------------------------------------
__global__ void row_stats(const float* __restrict__ logits, float* __restrict__ maxv,
                          float* __restrict__ sumv) {
  __shared__ float red[8];
  const int b = blockIdx.x, tid = threadIdx.x;   // 256 threads
  const float* lr = logits + (size_t)b * VV;
  float m = -1e30f;
  for (int v = tid; v < VV; v += 256) m = fmaxf(m, lr[v]);
  m = block_reduce_256(m, red, true);
  float s = 0.f;
  for (int v = tid; v < VV; v += 256) s += __expf(lr[v] - m);
  s = block_reduce_256(s, red, false);
  if (tid == 0) { maxv[b] = m; sumv[b] = s; }
}

// ---------------------------------------------------------------------------
// Final distribution: gen softmax * ptr, scatter-add copy mass, renormalize.
// One block per batch row -> scatter atomics stay within one row/block.
// ---------------------------------------------------------------------------
__global__ void finalize(const float* __restrict__ logits, const float* __restrict__ maxv,
                         const float* __restrict__ sumv, const float* __restrict__ ptrbuf,
                         const int* __restrict__ kg_ext, const float* __restrict__ val,
                         float* __restrict__ out, int t) {
  __shared__ float red[8];
  const int b = blockIdx.x, tid = threadIdx.x;   // 256 threads
  float* orow = out + ((size_t)b * T_ + t) * VX_;
  const float* lr = logits + (size_t)b * VV;
  const float mx = maxv[b];
  const float sc = ptrbuf[b] / sumv[b];
  for (int v = tid; v < VX_; v += 256)
    orow[v] = (v < VV) ? __expf(lr[v] - mx) * sc : 0.f;
  __threadfence();
  __syncthreads();
  for (int i = tid; i < KK * LL; i += 256) {
    int idx = kg_ext[b * (KK * LL) + i];
    atomicAdd(&orow[idx], val[b * (KK * LL) + i]);
  }
  __threadfence();
  __syncthreads();
  float s = 0.f;
  for (int v = tid; v < VX_; v += 256) s += orow[v];
  s = block_reduce_256(s, red, false);
  float iv = 1.f / s;
  for (int v = tid; v < VX_; v += 256) orow[v] *= iv;
}

// ---------------------------------------------------------------------------
// Host launcher
// ---------------------------------------------------------------------------
extern "C" void kernel_launch(void* const* d_in, const int* in_sizes, int n_in,
                              void* d_out, int out_size, void* d_ws, size_t ws_size,
                              hipStream_t stream) {
  (void)in_sizes; (void)n_in; (void)out_size; (void)ws_size;

  const int*   tgt        = (const int*)d_in[0];
  const float* init_h     = (const float*)d_in[1];
  const float* src_out    = (const float*)d_in[2];
  const float* src_mask   = (const float*)d_in[3];
  const float* kg_hid     = (const float*)d_in[4];
  const float* kg_fus     = (const float*)d_in[5];
  const float* kg_outputs = (const float*)d_in[6];
  const float* kg_mask    = (const float*)d_in[7];
  const float* kg_pad     = (const float*)d_in[8];
  /* d_in[9] global_kg_dist: unused by reference */
  const int*   kg_ext     = (const int*)d_in[10];
  /* d_in[11] extra_zero: zeros */
  const float* embedding  = (const float*)d_in[12];
  const float* W_ih       = (const float*)d_in[13];
  const float* W_hh       = (const float*)d_in[14];
  const float* b_ih       = (const float*)d_in[15];
  const float* b_hh       = (const float*)d_in[16];
  const float* Wq_c       = (const float*)d_in[17];
  const float* Wv_c       = (const float*)d_in[18];
  const float* bv_c       = (const float*)d_in[19];
  const float* Vw_c       = (const float*)d_in[20];
  const float* bV_c       = (const float*)d_in[21];
  const float* Wq_k       = (const float*)d_in[22];
  const float* Wv_k       = (const float*)d_in[23];
  const float* bv_k       = (const float*)d_in[24];
  const float* Vw_k       = (const float*)d_in[25];
  const float* bV_k       = (const float*)d_in[26];
  const float* W1         = (const float*)d_in[27];
  const float* b1         = (const float*)d_in[28];
  const float* W2         = (const float*)d_in[29];
  const float* b2         = (const float*)d_in[30];
  const float* Wptr       = (const float*)d_in[31];
  const float* bptr       = (const float*)d_in[32];
  const float* Wcpy       = (const float*)d_in[33];
  const float* bcpy       = (const float*)d_in[34];

  // Workspace carve-up (~70 MB), 256B-aligned chunks
  char* ws = (char*)d_ws;
  size_t off = 0;
  auto alloc = [&](size_t bytes) -> char* {
    char* p = ws + off;
    off += (bytes + 255) & ~(size_t)255;
    return p;
  };
  // f16 weights / static tensors
  _Float16* W_ih_h   = (_Float16*)alloc((size_t)1536 * 1536 * 2);
  _Float16* W_hh_h   = (_Float16*)alloc((size_t)1536 * 512 * 2);
  _Float16* W1_h     = (_Float16*)alloc((size_t)512 * 2048 * 2);
  _Float16* W2_h     = (_Float16*)alloc((size_t)VV * 512 * 2);
  _Float16* Wcpy_h   = (_Float16*)alloc((size_t)512 * 512 * 2);
  _Float16* Wv_c_h   = (_Float16*)alloc((size_t)512 * 512 * 2);
  _Float16* Wv_k_h   = (_Float16*)alloc((size_t)512 * 512 * 2);
  _Float16* kgout_h  = (_Float16*)alloc((size_t)B_ * KK * LL * HH * 2);
  _Float16* src_o_h  = (_Float16*)alloc((size_t)B_ * S_ * HH * 2);
  _Float16* kg_hid_h = (_Float16*)alloc((size_t)B_ * KK * HH * 2);
  // f32 state / intermediates
  float* pv_src = (float*)alloc((size_t)B_ * S_ * HH * 4);
  float* pv_kg  = (float*)alloc((size_t)B_ * KK * HH * 4);
  float* hbuf   = (float*)alloc((size_t)B_ * HH * 4);
  float* cbuf   = (float*)alloc((size_t)B_ * HH * 4);
  float* kbuf   = (float*)alloc((size_t)B_ * HH * 4);
  float* qbuf   = (float*)alloc((size_t)B_ * HH * 4);
  float* xbuf   = (float*)alloc((size_t)B_ * 1536 * 4);
  float* gibuf  = (float*)alloc((size_t)B_ * 1536 * 4);
  float* ghbuf  = (float*)alloc((size_t)B_ * 1536 * 4);
  float* kwbuf  = (float*)alloc((size_t)B_ * KK * 4);
  float* ptrb   = (float*)alloc((size_t)B_ * 4);
  float* maxb   = (float*)alloc((size_t)B_ * 4);
  float* sumb   = (float*)alloc((size_t)B_ * 4);
  float* valb   = (float*)alloc((size_t)B_ * KK * LL * 4);
  float* logit  = (float*)alloc((size_t)B_ * VV * 4);
  // f16 activations (GEMM A fragments load straight as b128)
  _Float16* hbuf_h = (_Float16*)alloc((size_t)B_ * HH * 2);
  _Float16* sbuf_h = (_Float16*)alloc((size_t)B_ * HH * 2);
  _Float16* xbuf_h = (_Float16*)alloc((size_t)B_ * 1536 * 2);
  _Float16* scat_h = (_Float16*)alloc((size_t)B_ * 2048 * 2);
  _Float16* ff_h   = (_Float16*)alloc((size_t)B_ * HH * 2);

  // ---- one-time prep: f16 copies (stay resident in 192MB L2) ----
  cvt_f16<<<512, 256, 0, stream>>>(W_ih, W_ih_h, (size_t)1536 * 1536);
  cvt_f16<<<512, 256, 0, stream>>>(W_hh, W_hh_h, (size_t)1536 * 512);
  cvt_f16<<<512, 256, 0, stream>>>(W1,   W1_h,   (size_t)512 * 2048);
  cvt_f16<<<1024, 256, 0, stream>>>(W2,  W2_h,   (size_t)VV * 512);
  cvt_f16<<<256, 256, 0, stream>>>(Wcpy, Wcpy_h, (size_t)512 * 512);
  cvt_f16<<<256, 256, 0, stream>>>(Wv_c, Wv_c_h, (size_t)512 * 512);
  cvt_f16<<<256, 256, 0, stream>>>(Wv_k, Wv_k_h, (size_t)512 * 512);
  cvt_f16<<<1024, 256, 0, stream>>>(kg_outputs, kgout_h, (size_t)B_ * KK * LL * HH);
  cvt_f16<<<1024, 256, 0, stream>>>(src_out, src_o_h, (size_t)B_ * S_ * HH);
  cvt_f16<<<64, 256, 0, stream>>>(kg_hid, kg_hid_h, (size_t)B_ * KK * HH);
  cvt_f16<<<32, 256, 0, stream>>>(init_h, hbuf_h, (size_t)B_ * HH);

  auto blocksFor = [](int M, int N) { return ((M / 16) * (N / 16) + 7) / 8; };

  // precomputed attention value projections (WMMA)
  gemm_h<<<blocksFor(B_ * S_, HH), 256, 0, stream>>>(src_o_h, Wv_c_h, bv_c,
                                                     pv_src, nullptr, B_ * S_, HH, HH);
  gemm_h<<<blocksFor(B_ * KK, HH), 256, 0, stream>>>(kg_hid_h, Wv_k_h, bv_k,
                                                     pv_kg, nullptr, B_ * KK, HH, HH);

  // h0 (f32 copy; f16 via cvt above)
  hipMemcpyAsync(hbuf, init_h, (size_t)B_ * HH * 4, hipMemcpyDeviceToDevice, stream);

  // ---- sequential decode: T=32 stream-ordered steps ----
  for (int t = 0; t < T_; ++t) {
    attend_kernel<<<B_, 256, 0, stream>>>(hbuf, Wq_c, Vw_c, bV_c, pv_src, src_out, src_mask,
                                          Wq_k, Vw_k, bV_k, pv_kg, kg_hid, kg_mask,
                                          cbuf, kbuf, kwbuf);
    build_x<<<B_, 256, 0, stream>>>(tgt, embedding, cbuf, kbuf, xbuf, xbuf_h, t);
    gemm_h<<<blocksFor(B_, 1536), 256, 0, stream>>>(xbuf_h, W_ih_h, b_ih,
                                                    gibuf, nullptr, B_, 1536, 1536);
    gemm_h<<<blocksFor(B_, 1536), 256, 0, stream>>>(hbuf_h, W_hh_h, b_hh,
                                                    ghbuf, nullptr, B_, 1536, 512);
    gru_combine<<<B_, 256, 0, stream>>>(gibuf, ghbuf, hbuf, hbuf_h, cbuf, kbuf, xbuf,
                                        kg_fus, Wptr, bptr, sbuf_h, scat_h, ptrb);
    gemm_h<<<blocksFor(B_, 512), 256, 0, stream>>>(sbuf_h, Wcpy_h, bcpy,
                                                   qbuf, nullptr, B_, 512, 512);
    gemm_h<<<blocksFor(B_, 512), 256, 0, stream>>>(scat_h, W1_h, b1,
                                                   nullptr, ff_h, B_, 512, 2048);
    gemm_h<<<blocksFor(B_, VV), 256, 0, stream>>>(ff_h, W2_h, b2,
                                                  logit, nullptr, B_, VV, 512);
    copy_scores<<<B_ * KK, 64, 0, stream>>>(qbuf, kgout_h, kg_pad, kwbuf, ptrb, valb);
    row_stats<<<B_, 256, 0, stream>>>(logit, maxb, sumb);
    finalize<<<B_, 256, 0, stream>>>(logit, maxb, sumb, ptrb, kg_ext, valb, (float*)d_out, t);
  }
}
